// ItemEncoder_49881750176202
// MI455X (gfx1250) — matrix operand order repaired and checked by reading, not compiled, not run on hardware
//
#include <hip/hip_runtime.h>
#include <hip/hip_bf16.h>

typedef __attribute__((ext_vector_type(16))) _Float16 v16h;
typedef __attribute__((ext_vector_type(8)))  float    v8f;

#define N_ITEM_TYPE 18
#define N_TYPE_HID  32
#define N_FEAT      11
#define KDIM        43      // 32 + 11
#define KPAD        64      // pad to 2 x 32 (two f16 WMMA k-steps)
#define NHID        128
#define NITEM       128

// One block handles one (bs, na) pair: a 128x64 (M x Kpad) f16 tile times
// a 64x128 (Kpad x N) f16 weight tile -> 128x128 f32, then bias+ReLU+max over M.
__global__ __launch_bounds__(256) void item_encoder_kernel(
    const int*   __restrict__ item_type,   // [BS*NA*NI] (harness: integer -> int*)
    const float* __restrict__ item,        // [BS*NA*NI, 11]
    const float* __restrict__ emb,         // [18, 32]
    const float* __restrict__ W,           // [43, 128]
    const float* __restrict__ bias,        // [128]
    float*       __restrict__ out)         // [BS*NA, 128]
{
    __shared__ _Float16 sh_h[NITEM * KPAD];     // activations, [row][k]   16 KB
    __shared__ _Float16 sh_wt[NHID * KPAD];     // W transposed, [n][k]    16 KB
    __shared__ _Float16 sh_emb[N_ITEM_TYPE * N_TYPE_HID];
    __shared__ float    sh_bias[NHID];
    __shared__ float    sh_red[16 * NHID];      // 8 waves x 2 lane-groups   8 KB

    const int t    = threadIdx.x;               // 0..255
    const long base = blockIdx.x;               // flat (b*NA + a), 0..4095

    // ---- stage embedding table, W^T (zero-padded K), bias into LDS ----
    for (int i = t; i < N_ITEM_TYPE * N_TYPE_HID; i += 256)
        sh_emb[i] = (_Float16)emb[i];
    if (t < NHID) {
        const int n = t;
        #pragma unroll 4
        for (int k = 0; k < KPAD; ++k)
            sh_wt[n * KPAD + k] = (_Float16)((k < KDIM) ? W[k * NHID + n] : 0.f);
        sh_bias[n] = bias[n];
    }
    __syncthreads();

    // ---- build the 128x64 activation tile: [emb(type) | feat | 0-pad] ----
    {
        const int r    = t >> 1;                // row 0..127
        const int half = t & 1;                 // which 32 columns of K
        if (half == 0) {
            const int ty = item_type[base * NITEM + r];
            const _Float16* e = sh_emb + ty * N_TYPE_HID;
            #pragma unroll
            for (int k = 0; k < N_TYPE_HID; ++k)
                sh_h[r * KPAD + k] = e[k];
        } else {
            const float* ip = item + (base * NITEM + r) * N_FEAT;
            #pragma unroll
            for (int k = 0; k < N_FEAT; ++k)
                sh_h[r * KPAD + N_TYPE_HID + k] = (_Float16)ip[k];
            #pragma unroll
            for (int k = N_FEAT; k < 32; ++k)
                sh_h[r * KPAD + N_TYPE_HID + k] = (_Float16)0.f;
        }
    }
    __syncthreads();

    // ---- WMMA GEMM: wave w owns M-tile [16w, 16w+16) ----
    const int wave = t >> 5;
    const int lane = t & 31;
    const int lrow = lane & 15;                 // M within tile / N within tile
    const int grp  = lane >> 4;                 // lane group 0/1

    // A fragments (16x32 f16): lanes 0-15 -> Kbase 0, lanes 16-31 -> Kbase 16.
    // elems 0..15 -> K = kb+0..15 (contiguous, matches B)
    const _Float16* hrow = sh_h + (wave * 16 + lrow) * KPAD;
    const int kb = grp * 16;
    v16h a0, a1;
    #pragma unroll
    for (int e = 0; e < 8; ++e) {
        a0[e]     = hrow[kb + e];
        a0[8 + e] = hrow[kb + 8 + e];
        a1[e]     = hrow[32 + kb + e];
        a1[8 + e] = hrow[32 + kb + 8 + e];
    }

    #pragma unroll
    for (int nt = 0; nt < 8; ++nt) {
        const int n = nt * 16 + lrow;
        // B fragment (32x16 f16): lane group = 16 K-rows, elem = K offset;
        // W^T layout makes this one contiguous 32-byte LDS read.
        const _Float16* wp = sh_wt + n * KPAD + grp * 16;
        v16h b0 = *(const v16h*)(wp);           // K 0..31 step
        v16h b1 = *(const v16h*)(wp + 32);      // K 32..63 step

        v8f acc = {};
        acc = __builtin_amdgcn_wmma_f32_16x16x32_f16(
                  false, a0, false, b0, (short)0, acc, false, false);
        acc = __builtin_amdgcn_wmma_f32_16x16x32_f16(
                  false, a1, false, b1, (short)0, acc, false, false);

        // bias + ReLU + max over this lane's 8 M rows (C/D: VGPR r -> M = 8*grp + r)
        const float bn = sh_bias[n];
        float mx = 0.f;                          // ReLU outputs are >= 0
        #pragma unroll
        for (int i = 0; i < 8; ++i) {
            float v = acc[i] + bn;
            mx = fmaxf(mx, fmaxf(v, 0.f));
        }
        sh_red[(wave * 2 + grp) * NHID + n] = mx;
    }
    __syncthreads();

    // ---- final max over 16 partials per column ----
    if (t < NHID) {
        float mx = 0.f;
        #pragma unroll
        for (int j = 0; j < 16; ++j)
            mx = fmaxf(mx, sh_red[j * NHID + t]);
        out[base * NHID + t] = mx;
    }
}

extern "C" void kernel_launch(void* const* d_in, const int* in_sizes, int n_in,
                              void* d_out, int out_size, void* d_ws, size_t ws_size,
                              hipStream_t stream) {
    const int*   item_type = (const int*)  d_in[0];
    const float* item      = (const float*)d_in[1];
    const float* emb       = (const float*)d_in[2];
    const float* W         = (const float*)d_in[3];
    const float* bias      = (const float*)d_in[4];
    float*       out       = (float*)d_out;

    const int n_blocks = 32 * 128;   // BS * NA
    item_encoder_kernel<<<dim3(n_blocks), dim3(256), 0, stream>>>(
        item_type, item, emb, W, bias, out);
}
